// BlockDrop_31610959299228
// MI455X (gfx1250) — compile-verified, matrix operand order checked by reading, not run on hardware
//
#include <hip/hip_runtime.h>
#include <hip/hip_bf16.h>

// ---------------------------------------------------------------------------
// BlockDrop (gated ResNet-32 + policy ResNet) for gfx1250 (MI455X).
// Implicit-GEMM conv3x3 via v_wmma_f32_16x16x32_f16 (f32 accum).
// Activations: zero-halo padded NHWC f16  [N][H+2][W+2][C]  -> no bounds
// checks in the K loop. Weights repacked to [Cout][Kpad] f16, K=(ky,kx,ci),
// Kpad=roundup(9*Cin,32). CIN templated -> fully unrolled K loop where every
// fragment load is a global_load_b128 at an immediate offset. NT=Cout/16
// output tiles per wave reuse each A fragment across NT WMMAs.
// ---------------------------------------------------------------------------

typedef _Float16 half8  __attribute__((ext_vector_type(8)));
typedef _Float16 v16h   __attribute__((ext_vector_type(16)));
typedef float    v8f    __attribute__((ext_vector_type(8)));

#define NIMG 1024

// ---------------- weight repack: OIHW f32 -> [O][Kpad] f16 -----------------
__global__ void conv_w_repack(const float* __restrict__ w, _Float16* __restrict__ out,
                              int O, int I, int Ceff, int Kpad) {
  int idx = blockIdx.x * 256 + threadIdx.x;
  if (idx >= O * Kpad) return;
  int k = idx % Kpad, o = idx / Kpad;
  float v = 0.f;
  if (k < 9 * Ceff) {
    int pos = k / Ceff, ci = k % Ceff;
    int ky = pos / 3, kx = pos - ky * 3;
    if (ci < I) v = w[(((size_t)o * I + ci) * 3 + ky) * 3 + kx];
  }
  out[idx] = (_Float16)v;
}

// ---- input: NCHW f32 (C=3) -> padded NHWC f16 [N][34][34][16], halo=0 -----
__global__ void pad_input(const float* __restrict__ in, _Float16* __restrict__ out, int total) {
  int idx = blockIdx.x * 256 + threadIdx.x;
  if (idx >= total) return;
  int c = idx & 15, pix = idx >> 4;
  int x = pix % 34; int t = pix / 34; int y = t % 34; int n = t / 34;
  float v = 0.f;
  if (c < 3 && y >= 1 && y <= 32 && x >= 1 && x <= 32)
    v = in[(((size_t)n * 3 + c) * 32 + (y - 1)) * 32 + (x - 1)];
  out[idx] = (_Float16)v;
}

// ------------------- bulk zero fill (16B granules) -------------------------
__global__ void zero_fill(uint4* __restrict__ p, size_t n16) {
  size_t idx = (size_t)blockIdx.x * 256 + threadIdx.x;
  if (idx < n16) p[idx] = make_uint4(0u, 0u, 0u, 0u);
}

// --------- DownsampleB: avgpool2x2 + zero-pad channels, padded layouts -----
__global__ void downsample_b_k(const _Float16* __restrict__ in, _Float16* __restrict__ out,
                               int Hi, int Wi, int C, int total) {
  int idx = blockIdx.x * 256 + threadIdx.x;
  if (idx >= total) return;
  int C2 = 2 * C;
  int c = idx % C2, pix = idx / C2;
  int Ho = Hi / 2, Wo = Wi / 2;
  int x = pix % Wo; int t = pix / Wo; int y = t % Ho; int n = t / Ho;
  float v = 0.f;
  if (c < C) {
    const _Float16* b0 = in + (((size_t)n * (Hi + 2) + 2 * y + 1) * (Wi + 2) + 2 * x + 1) * C + c;
    size_t rs = (size_t)(Wi + 2) * C;
    v = 0.25f * ((float)b0[0] + (float)b0[C] + (float)b0[rs] + (float)b0[rs + C]);
  }
  out[(((size_t)n * (Ho + 2) + y + 1) * (Wo + 2) + x + 1) * C2 + c] = (_Float16)v;
}

// ------------- fused conv3x3 + BN (+relu / +gated residual) ----------------
// One wave computes a 16(M) x (NT*16)(N) tile. Fully unrolled K loop: all
// fragment loads are b128 at compile-time immediate offsets from 3 row ptrs.
template <int CIN, int NT>
__global__ __launch_bounds__(256)
void conv3x3_wmma(const _Float16* __restrict__ in, const _Float16* __restrict__ wt,
                  const float* __restrict__ bng, const float* __restrict__ bnb,
                  const float* __restrict__ bnm, const float* __restrict__ bnv,
                  _Float16* __restrict__ out,
                  const _Float16* __restrict__ residual,
                  const float* __restrict__ policy, int t,
                  int Hin, int Win, int stride, int Hout, int Wout,
                  int lHW, int lW, int totalTiles) {
  constexpr int COUT = NT * 16;
  constexpr int K9   = 9 * CIN;
  constexpr int KPAD = (K9 + 31) & ~31;
  constexpr int NCH  = KPAD / 32;

  int wave = blockIdx.x * 8 + (threadIdx.x >> 5);
  if (wave >= totalTiles) return;           // wave-uniform: EXEC stays all-ones
  int lane  = threadIdx.x & 31;
  int lrow = lane & 15, lhalf = lane >> 4, kb = lhalf << 3;

  // A-fragment row (output position) for this lane
  int p   = wave * 16 + lrow;
  int img = p >> lHW;
  int rem = p & ((1 << lHW) - 1);
  int oy  = rem >> lW;
  int ox  = rem & ((1 << lW) - 1);
  int Wp = Win + 2;

  const _Float16* arow[3];
#pragma unroll
  for (int ky = 0; ky < 3; ++ky)
    arow[ky] = in + (((size_t)img * (Hin + 2) + oy * stride + ky) * Wp + ox * stride) * CIN + kb;

  const _Float16* wrow[NT];
#pragma unroll
  for (int nt = 0; nt < NT; ++nt)
    wrow[nt] = wt + (size_t)(nt * 16 + lrow) * KPAD + kb;

  v8f acc[NT] = {};

#pragma unroll
  for (int kc = 0; kc < NCH; ++kc) {
    v16h afrag;
#pragma unroll
    for (int r = 0; r < 2; ++r) {
      half8 arun = {};
      const int kk = kc * 32 + r * 16;       // compile-time under unroll
      if (kk < K9) {                          // folds at compile time
        const int pos = kk / CIN;             // kernel tap index (0..8)
        const int cb  = kk % CIN;             // channel base within tap
        const int ky = pos / 3, kx = pos - ky * 3;
        arun = *(const half8*)(arow[ky] + kx * CIN + cb);
      }
#pragma unroll
      for (int j = 0; j < 8; ++j) afrag[r * 8 + j] = arun[j];
    }
#pragma unroll
    for (int nt = 0; nt < NT; ++nt) {
      half8 b0 = *(const half8*)(wrow[nt] + kc * 32);
      half8 b1 = *(const half8*)(wrow[nt] + kc * 32 + 16);
      v16h bfrag;
#pragma unroll
      for (int j = 0; j < 8; ++j) { bfrag[j] = b0[j]; bfrag[8 + j] = b1[j]; }
      acc[nt] = __builtin_amdgcn_wmma_f32_16x16x32_f16(false, afrag, false, bfrag,
                                                       (short)0, acc[nt], false, false);
    }
  }

  // ---- epilogue: BN (+relu) or (+residual, relu, policy blend) ----
  float scale[NT], bias[NT];
#pragma unroll
  for (int nt = 0; nt < NT; ++nt) {
    int n = nt * 16 + lrow;
    float sc = bng[n] * rsqrtf(bnv[n] + 1e-5f);
    scale[nt] = sc;
    bias[nt]  = bnb[n] - bnm[n] * sc;
  }
  int Wop = Wout + 2;
#pragma unroll
  for (int r = 0; r < 8; ++r) {
    int q  = wave * 16 + r + (lhalf << 3);
    int qi = q >> lHW;
    int qr = q & ((1 << lHW) - 1);
    int qy = qr >> lW;
    int qx = qr & ((1 << lW) - 1);
    size_t pix = (((size_t)qi * (Hout + 2) + qy + 1) * Wop + qx + 1) * COUT;
    float apol = 1.f, onem = 0.f;
    if (policy) { apol = policy[t * NIMG + qi]; onem = 1.f - apol; }
#pragma unroll
    for (int nt = 0; nt < NT; ++nt) {
      int n = nt * 16 + lrow;
      float val = acc[nt][r] * scale[nt] + bias[nt];
      size_t oidx = pix + n;
      if (residual) {
        float res = (float)residual[oidx];
        float fx  = fmaxf(res + val, 0.f);
        out[oidx] = (_Float16)(policy ? (apol * fx + onem * res) : fx);
      } else {
        out[oidx] = (_Float16)fmaxf(val, 0.f);
      }
    }
  }
}

// --------- agent head: avgpool8x8 -> sigmoid(logit) -> binarize ------------
__global__ __launch_bounds__(64)
void policy_head(const _Float16* __restrict__ x, const float* __restrict__ lw,
                 const float* __restrict__ lb, float* __restrict__ pol) {
  __shared__ float y[64];
  int img = blockIdx.x, c = threadIdx.x;
  float s = 0.f;
  for (int py = 0; py < 8; ++py)
    for (int px = 0; px < 8; ++px)
      s += (float)x[(((size_t)img * 10 + py + 1) * 10 + px + 1) * 64 + c];
  y[c] = s * (1.f / 64.f);
  __syncthreads();
  if (c < 15) {
    float acc = lb[c];
    for (int k = 0; k < 64; ++k) acc += y[k] * lw[c * 64 + k];
    float prob = 1.f / (1.f + __expf(-acc));
    pol[c * NIMG + img] = (prob < 0.5f) ? 0.f : 1.f;
  }
}

// ---------------- final head: avgpool8x8 -> fc(64->10), f32 out ------------
__global__ __launch_bounds__(64)
void fc_head(const _Float16* __restrict__ x, const float* __restrict__ fw,
             const float* __restrict__ fb, float* __restrict__ out) {
  __shared__ float y[64];
  int img = blockIdx.x, c = threadIdx.x;
  float s = 0.f;
  for (int py = 0; py < 8; ++py)
    for (int px = 0; px < 8; ++px)
      s += (float)x[(((size_t)img * 10 + py + 1) * 10 + px + 1) * 64 + c];
  y[c] = s * (1.f / 64.f);
  __syncthreads();
  if (c < 10) {
    float acc = fb[c];
    for (int k = 0; k < 64; ++k) acc += y[k] * fw[c * 64 + k];
    out[(size_t)img * 10 + c] = acc;
  }
}

// ---------------------------------------------------------------------------
// Host orchestration
// ---------------------------------------------------------------------------
static int ilog2i(int v) { int l = 0; while ((1 << l) < v) ++l; return l; }

static void run_conv(hipStream_t stream,
                     const _Float16* in, _Float16* out, const _Float16* wt,
                     const float* g, const float* b, const float* m, const float* v,
                     int Hin, int Cin, int Cout, int stride, int Hout,
                     const _Float16* residual, const float* policy, int t) {
  int Wout = Hout, Win = Hin;
  int lHW = ilog2i(Hout * Wout), lW = ilog2i(Wout);
  int tilesM = (NIMG * Hout * Wout) / 16;
  int blocks = (tilesM + 7) / 8;
#define LAUNCH(CI, NT)                                                          \
  conv3x3_wmma<CI, NT><<<blocks, 256, 0, stream>>>(in, wt, g, b, m, v, out,     \
      residual, policy, t, Hin, Win, stride, Hout, Wout, lHW, lW, tilesM)
  if      (Cin == 16 && Cout == 16) LAUNCH(16, 1);
  else if (Cin == 16 && Cout == 32) LAUNCH(16, 2);
  else if (Cin == 32 && Cout == 32) LAUNCH(32, 2);
  else if (Cin == 32 && Cout == 64) LAUNCH(32, 4);
  else                              LAUNCH(64, 4);
#undef LAUNCH
}

// Flattened input index map (setup_inputs() dict insertion order, recursive):
//   0: inputs
//   rnet: 1 conv1; 2..5 bn1{g,b,m,v}; blocks[s][b] base=6+(s*5+b)*10:
//         +0 c1, +1..4 b1{g,b,m,v}, +5 c2, +6..9 b2{g,b,m,v}; 156 fc_w; 157 fc_b
//   agent: 158 conv1; 159..162 bn1; blocks[s] base=163+s*10; 193 logit_w; 194 logit_b
#define RN_CONV1 1
#define RN_BN1   2
#define RN_BLK(s,b) (6 + ((s)*5 + (b))*10)
#define RN_FC_W  156
#define RN_FC_B  157
#define AG_CONV1 158
#define AG_BN1   159
#define AG_BLK(s) (163 + (s)*10)
#define AG_LOGIT_W 193
#define AG_LOGIT_B 194

extern "C" void kernel_launch(void* const* d_in, const int* in_sizes, int n_in,
                              void* d_out, int out_size, void* d_ws, size_t ws_size,
                              hipStream_t stream) {
  (void)in_sizes; (void)n_in; (void)out_size; (void)ws_size;
  auto F = [&](int i) { return (const float*)d_in[i]; };

  // ---- workspace carve-up (all sizes are multiples of 256B) ----
  char* ws = (char*)d_ws;
  size_t off = 0;
  auto carve = [&](size_t bytes) {
    void* p = ws + off;
    off = (off + bytes + 255) & ~(size_t)255;
    return p;
  };
  const size_t ACTP = (size_t)NIMG * 34 * 34 * 16 * 2;       // padded 32x32x16 acts
  const size_t RESP = (size_t)NIMG * 18 * 18 * 32 * 2;       // padded 16x16x32 acts
  _Float16* inPad = (_Float16*)carve(ACTP);
  _Float16* X     = (_Float16*)carve(ACTP);
  _Float16* Y     = (_Float16*)carve(ACTP);
  _Float16* Obuf  = (_Float16*)carve(ACTP);
  _Float16* R     = (_Float16*)carve(RESP);
  float*    pol   = (float*)carve((size_t)15 * NIMG * 4);
  _Float16* WB    = (_Float16*)carve((size_t)38 * 36864 * 2); // 38 conv weight slots

  // zero X/Y/Obuf/R once so all halos are 0 (convs only write interiors)
  {
    size_t n16 = (3 * ACTP + RESP) / 16;
    zero_fill<<<(int)((n16 + 255) / 256), 256, 0, stream>>>((uint4*)X, n16);
  }

  int wslot = 0;
  auto repack = [&](int idx, int I, int Ceff, int O) -> const _Float16* {
    int Kpad = ((9 * Ceff + 31) / 32) * 32;
    _Float16* dst = WB + (size_t)(wslot++) * 36864;
    int tot = O * Kpad;
    conv_w_repack<<<(tot + 255) / 256, 256, 0, stream>>>(F(idx), dst, O, I, Ceff, Kpad);
    return dst;
  };
  auto downsample = [&](const _Float16* src, _Float16* dst, int Hi, int C) {
    int tot = NIMG * (Hi / 2) * (Hi / 2) * 2 * C;
    downsample_b_k<<<(tot + 255) / 256, 256, 0, stream>>>(src, dst, Hi, Hi, C, tot);
  };

  // stage geometry
  const int C_IN[3]  = {16, 16, 32};
  const int C_OUT[3] = {16, 32, 64};
  const int H_IN[3]  = {32, 32, 16};
  const int H_OUT[3] = {32, 16, 8};
  const int STR[3]   = {1, 2, 2};

  // ---- input -> padded NHWC f16 (C padded 3->16, halo zeroed) ----
  {
    int tot = NIMG * 34 * 34 * 16;
    pad_input<<<(tot + 255) / 256, 256, 0, stream>>>(F(0), inPad, tot);
  }

  // =================== agent network (FlatResNet32Policy [1,1,1]) ==========
  {
    const _Float16* w = repack(AG_CONV1, 3, 16, 16);
    run_conv(stream, inPad, X, w, F(AG_BN1), F(AG_BN1 + 1), F(AG_BN1 + 2), F(AG_BN1 + 3),
             32, 16, 16, 1, 32, nullptr, nullptr, 0);
    for (int s = 0; s < 3; ++s) {
      int bi = AG_BLK(s);
      int Ci = C_IN[s], Co = C_OUT[s], Hi = H_IN[s], Ho = H_OUT[s], st = STR[s];
      const _Float16* res = X;
      if (s > 0) { downsample(X, R, Hi, Ci); res = R; }
      const _Float16* w1 = repack(bi + 0, Ci, Ci, Co);
      run_conv(stream, X, Obuf, w1, F(bi + 1), F(bi + 2), F(bi + 3), F(bi + 4),
               Hi, Ci, Co, st, Ho, nullptr, nullptr, 0);
      const _Float16* w2 = repack(bi + 5, Co, Co, Co);
      run_conv(stream, Obuf, Y, w2, F(bi + 6), F(bi + 7), F(bi + 8), F(bi + 9),
               Ho, Co, Co, 1, Ho, res, nullptr, 0);   // policy=null -> x=relu(res+fx)
      _Float16* tmp = X; X = Y; Y = tmp;
    }
    policy_head<<<NIMG, 64, 0, stream>>>(X, F(AG_LOGIT_W), F(AG_LOGIT_B), pol);
  }

  // =================== main network (ResNet-32, gated) =====================
  {
    const _Float16* w = repack(RN_CONV1, 3, 16, 16);
    run_conv(stream, inPad, X, w, F(RN_BN1), F(RN_BN1 + 1), F(RN_BN1 + 2), F(RN_BN1 + 3),
             32, 16, 16, 1, 32, nullptr, nullptr, 0);
    int t = 0;
    for (int s = 0; s < 3; ++s) {
      for (int b = 0; b < 5; ++b) {
        int bi = RN_BLK(s, b);
        int Ci = (b == 0) ? C_IN[s] : C_OUT[s];
        int Co = C_OUT[s];
        int Hi = (b == 0) ? H_IN[s] : H_OUT[s];
        int Ho = H_OUT[s];
        int st = (b == 0) ? STR[s] : 1;
        const _Float16* res = X;
        if (b == 0 && s > 0) { downsample(X, R, Hi, Ci); res = R; }
        const _Float16* w1 = repack(bi + 0, Ci, Ci, Co);
        run_conv(stream, X, Obuf, w1, F(bi + 1), F(bi + 2), F(bi + 3), F(bi + 4),
                 Hi, Ci, Co, st, Ho, nullptr, nullptr, 0);
        const _Float16* w2 = repack(bi + 5, Co, Co, Co);
        run_conv(stream, Obuf, Y, w2, F(bi + 6), F(bi + 7), F(bi + 8), F(bi + 9),
                 Ho, Co, Co, 1, Ho, res, pol, t);
        _Float16* tmp = X; X = Y; Y = tmp;
        ++t;
      }
    }
    fc_head<<<NIMG, 64, 0, stream>>>(X, F(RN_FC_W), F(RN_FC_B), (float*)d_out);
  }
}
